// GRU_30107720745592
// MI455X (gfx1250) — matrix operand order repaired and checked by reading, not compiled, not run on hardware
//
#include <hip/hip_runtime.h>
#include <hip/hip_bf16.h>

// ---------------------------------------------------------------------------
// Types for CDNA5 WMMA (wave32): D(f32 16x16) = A(bf16 16x32) * B(bf16 32x16) + C
// ---------------------------------------------------------------------------
typedef __bf16 bf16_t;
typedef __attribute__((ext_vector_type(16))) __bf16 v16bf;
typedef __attribute__((ext_vector_type(8)))  __bf16 v8bf;
typedef __attribute__((ext_vector_type(8)))  float  v8f;

// ---------------------------------------------------------------------------
// Async global -> LDS (CDNA5): tracked by ASYNCcnt, no VGPR round trip.
//   VDST = per-lane LDS byte address, VADDR = per-lane 64-bit global address.
// ---------------------------------------------------------------------------
static __device__ __forceinline__
void async_g2l_b128(uint32_t lds_addr, const bf16_t* gaddr) {
    asm volatile("global_load_async_to_lds_b128 %0, %1, off"
                 :: "v"(lds_addr), "v"(gaddr) : "memory");
}
static __device__ __forceinline__ void wait_async0() {
    asm volatile("s_wait_asynccnt 0x0" ::: "memory");
}

// Cooperative stage of a 64x64 bf16 tile (row-major, leading dim `ld`) into LDS
// (row-major, ld 64). 128 threads * 4 iters * 16B = 8 KB.
static __device__ __forceinline__
void stage_tile_async(bf16_t* dst, const bf16_t* src, int ld, int tid) {
#pragma unroll
    for (int i = 0; i < 4; ++i) {
        int seg = tid + i * 128;          // 512 x 16B segments
        int row = seg >> 3;               // 0..63
        int c0  = (seg & 7) * 8;          // element offset (8 bf16 = 16B)
        uint32_t laddr = (uint32_t)(uintptr_t)(dst + row * 64 + c0);
        async_g2l_b128(laddr, src + (size_t)row * ld + c0);
    }
}

// A fragment (16x32): lane r holds row r; half 0 -> K {0..7,16..23}, half 1 -> {8..15,24..31}
static __device__ __forceinline__
v16bf lds_frag_a(const bf16_t* tile, int row0, int kk, int lane) {
    const int r = lane & 15, half = lane >> 4;
    const bf16_t* p = tile + (row0 + r) * 64 + kk + half * 8;
    v8bf lo = *(const v8bf*)(p);
    v8bf hi = *(const v8bf*)(p + 16);
    return __builtin_shufflevector(lo, hi, 0,1,2,3,4,5,6,7,8,9,10,11,12,13,14,15);
}

// B fragment (32x16) from transposed weights tile: lane holds column n0+r;
// half 0 -> K {0..7,16..23}, half 1 -> {8..15,24..31}
static __device__ __forceinline__
v16bf lds_frag_b(const bf16_t* tile, int n0, int kk, int lane) {
    const int r = lane & 15, half = lane >> 4;
    const bf16_t* p = tile + (n0 + r) * 64 + kk + half * 8;
    v8bf lo = *(const v8bf*)(p);
    v8bf hi = *(const v8bf*)(p + 16);
    return __builtin_shufflevector(lo, hi, 0,1,2,3,4,5,6,7,8,9,10,11,12,13,14,15);
}

// ---------------------------------------------------------------------------
// Block-level 64(M) x 64(N) x K accumulate:
//   A:  block's 64 rows (M x K row-major, leading dim lda)
//   WTn: 64 rows of pre-transposed weights (N x K row-major) for this n-block
// LDS double-buffered 64x64 tiles; async DMA of chunk k+64 overlaps WMMA on k.
// Within a chunk, ALL ds_load fragments are issued before the first WMMA so the
// in-order DScnt lets the WMMA stream overlap DS latency (no s_wait_dscnt 0 per op).
// 4 waves; wave w owns columns w*16..w*16+15 -> acc[4] (M sub-tiles).
// ---------------------------------------------------------------------------
static __device__ __forceinline__
void gemm_block(const bf16_t* A, int lda, const bf16_t* WTn, int ldw,
                int K, int tid, v8f acc[4]) {
    __shared__ bf16_t sA[2][64 * 64];
    __shared__ bf16_t sB[2][64 * 64];
    const int lane = tid & 31, wave = tid >> 5;

    stage_tile_async(sA[0], A,   lda, tid);
    stage_tile_async(sB[0], WTn, ldw, tid);
    wait_async0();
    __syncthreads();

    int buf = 0;
    for (int k0 = 0; k0 < K; k0 += 64) {
        const int nk = k0 + 64;
        if (nk < K) {                       // prefetch next chunk while computing
            stage_tile_async(sA[buf ^ 1], A + nk,   lda, tid);
            stage_tile_async(sB[buf ^ 1], WTn + nk, ldw, tid);
        }
        const bf16_t* curA = sA[buf];
        const bf16_t* curB = sB[buf];

        // Phase 1: issue every fragment load for this 64-wide K chunk.
        v16bf bf[2];
        v16bf af[2][4];
#pragma unroll
        for (int kk = 0; kk < 2; ++kk) {
            bf[kk] = lds_frag_b(curB, wave * 16, kk * 32, lane);
#pragma unroll
            for (int mt = 0; mt < 4; ++mt)
                af[kk][mt] = lds_frag_a(curA, mt * 16, kk * 32, lane);
        }
        // Phase 2: back-to-back WMMAs; DS results drain in-order underneath.
#pragma unroll
        for (int kk = 0; kk < 2; ++kk)
#pragma unroll
            for (int mt = 0; mt < 4; ++mt)
                acc[mt] = __builtin_amdgcn_wmma_f32_16x16x32_bf16(
                    false, af[kk][mt], false, bf[kk], (short)0, acc[mt], false, false);

        if (nk < K) wait_async0();
        __syncthreads();
        buf ^= 1;
    }
}

// ---------------------------------------------------------------------------
// Big hoisted GEMM: Out(M x N, bf16) = A(M x K, bf16) @ WT(N x K, bf16)^T
// grid = (N/64, M/64), block = 128 threads
// ---------------------------------------------------------------------------
__global__ void gemm_bf16_kernel(const bf16_t* __restrict__ A, int lda,
                                 const bf16_t* __restrict__ WT, int K,
                                 bf16_t* __restrict__ Out, int ldo) {
    const int tid = threadIdx.x;
    const int lane = tid & 31, wave = tid >> 5;
    const int n0 = blockIdx.x * 64 + wave * 16;
    const size_t m0 = (size_t)blockIdx.y * 64;

    v8f acc[4] = {};
    gemm_block(A + m0 * lda, lda, WT + (size_t)blockIdx.x * 64 * K, K, K, tid, acc);

    const int r = lane & 15, half = lane >> 4;
    const int col = n0 + r;
#pragma unroll
    for (int mt = 0; mt < 4; ++mt)
#pragma unroll
        for (int i = 0; i < 8; ++i) {
            size_t row = m0 + mt * 16 + half * 8 + i;    // C/D layout: VGPR i, lane half
            Out[row * ldo + col] = (bf16_t)acc[mt][i];
        }
}

// ---------------------------------------------------------------------------
// GRU step, phase A: rz = h_prev @ Wh_rz + xw_rz + bx_rz + bh_rz
//   n < H  : r = sigmoid -> hr = h_prev * r  (bf16 matmul input for phase B)
//   n >= H : z = sigmoid -> zbuf
// M=64, N=2048, K=1024; grid = 32 blocks of 128 threads
// ---------------------------------------------------------------------------
__global__ void gru_step_rz(const bf16_t* __restrict__ Hprev_bf,   // 64 x 1024
                            const bf16_t* __restrict__ WhArzT,     // 2048 x 1024
                            const bf16_t* __restrict__ XWt,        // 64 x 3072 (cols 0..2047)
                            const float* __restrict__ bx_rz,
                            const float* __restrict__ bh_rz,
                            const float* __restrict__ hF,          // 64 x 1024 fp32 state
                            float* __restrict__ zbuf,              // 64 x 1024
                            bf16_t* __restrict__ hrbf) {           // 64 x 1024
    const int tid = threadIdx.x;
    const int lane = tid & 31, wave = tid >> 5;
    const int n0 = blockIdx.x * 64 + wave * 16;

    v8f acc[4] = {};
    gemm_block(Hprev_bf, 1024, WhArzT + (size_t)blockIdx.x * 64 * 1024, 1024, 1024, tid, acc);

    const int r = lane & 15, half = lane >> 4;
    const int col = n0 + r;
    const float brz = bx_rz[col] + bh_rz[col];
#pragma unroll
    for (int mt = 0; mt < 4; ++mt)
#pragma unroll
        for (int i = 0; i < 8; ++i) {
            int row = mt * 16 + half * 8 + i;
            float v = acc[mt][i] + (float)XWt[row * 3072 + col] + brz;
            float s = 1.0f / (1.0f + __expf(-v));
            if (col < 1024) {   // r gate -> h*r
                hrbf[row * 1024 + col] = (bf16_t)(hF[row * 1024 + col] * s);
            } else {            // z gate
                zbuf[row * 1024 + (col - 1024)] = s;
            }
        }
}

// ---------------------------------------------------------------------------
// GRU step, phase B: cand = tanh(hr @ Wh_h + xw_h + bx_h + bh_h)
//                    h_new = z*h + (1-z)*cand
// M=64, N=1024, K=1024; grid = 16 blocks of 128 threads
// ---------------------------------------------------------------------------
__global__ void gru_step_out(const bf16_t* __restrict__ hrbf,      // 64 x 1024
                             const bf16_t* __restrict__ WhBhT,     // 1024 x 1024
                             const bf16_t* __restrict__ XWt,       // 64 x 3072 (cols 2048..3071)
                             const float* __restrict__ bx_h,
                             const float* __restrict__ bh_h,
                             const float* __restrict__ zbuf,
                             float* __restrict__ hF,               // state, updated in place
                             bf16_t* __restrict__ hbf_next,        // bf16 copy for next step
                             float* __restrict__ outF32) {         // optional fp32 output row
    const int tid = threadIdx.x;
    const int lane = tid & 31, wave = tid >> 5;
    const int n0 = blockIdx.x * 64 + wave * 16;

    v8f acc[4] = {};
    gemm_block(hrbf, 1024, WhBhT + (size_t)blockIdx.x * 64 * 1024, 1024, 1024, tid, acc);

    const int r = lane & 15, half = lane >> 4;
    const int col = n0 + r;
    const float bh = bx_h[col] + bh_h[col];
#pragma unroll
    for (int mt = 0; mt < 4; ++mt)
#pragma unroll
        for (int i = 0; i < 8; ++i) {
            int row = mt * 16 + half * 8 + i;
            float v = acc[mt][i] + (float)XWt[row * 3072 + 2048 + col] + bh;
            float cand = tanhf(v);
            float z = zbuf[row * 1024 + col];
            float hnew = z * hF[row * 1024 + col] + (1.0f - z) * cand;
            hF[row * 1024 + col] = hnew;
            hbf_next[row * 1024 + col] = (bf16_t)hnew;
            if (outF32) outF32[row * 1024 + col] = hnew;
        }
}

// ---------------------------------------------------------------------------
// Prep kernels
// ---------------------------------------------------------------------------
__global__ void transpose_to_bf16(const float* __restrict__ in, bf16_t* __restrict__ outp,
                                  int K, int N) {  // in: K x N -> outp: N x K
    size_t idx = (size_t)blockIdx.x * blockDim.x + threadIdx.x;
    size_t total = (size_t)K * N;
    if (idx >= total) return;
    size_t n = idx / K, k = idx % K;
    outp[idx] = (bf16_t)in[k * N + n];
}

__global__ void cvt_to_bf16(const float* __restrict__ in, bf16_t* __restrict__ outp, size_t n) {
    size_t i = (size_t)blockIdx.x * blockDim.x + threadIdx.x;
    if (i < n) outp[i] = (bf16_t)in[i];
}

__global__ void zero_bf16(bf16_t* p, size_t n) {
    size_t i = (size_t)blockIdx.x * blockDim.x + threadIdx.x;
    if (i < n) p[i] = (bf16_t)0.0f;
}

__global__ void zero_f32(float* p, size_t n) {
    size_t i = (size_t)blockIdx.x * blockDim.x + threadIdx.x;
    if (i < n) p[i] = 0.0f;
}

__global__ void copy_f32(const float* __restrict__ in, float* __restrict__ outp, size_t n) {
    size_t i = (size_t)blockIdx.x * blockDim.x + threadIdx.x;
    if (i < n) outp[i] = in[i];
}

// ---------------------------------------------------------------------------
// Host orchestration
// ---------------------------------------------------------------------------
extern "C" void kernel_launch(void* const* d_in, const int* in_sizes, int n_in,
                              void* d_out, int out_size, void* d_ws, size_t ws_size,
                              hipStream_t stream) {
    (void)in_sizes; (void)n_in; (void)out_size; (void)ws_size;

    const float* X      = (const float*)d_in[0];
    const float* Wx_rz0 = (const float*)d_in[1];
    const float* bx_rz0 = (const float*)d_in[2];
    const float* Wh_rz0 = (const float*)d_in[3];
    const float* bh_rz0 = (const float*)d_in[4];
    const float* Wx_h0  = (const float*)d_in[5];
    const float* bx_h0  = (const float*)d_in[6];
    const float* Wh_h0  = (const float*)d_in[7];
    const float* bh_h0  = (const float*)d_in[8];
    const float* Wx_rz1 = (const float*)d_in[9];
    const float* bx_rz1 = (const float*)d_in[10];
    const float* Wh_rz1 = (const float*)d_in[11];
    const float* bh_rz1 = (const float*)d_in[12];
    const float* Wx_h1  = (const float*)d_in[13];
    const float* bx_h1  = (const float*)d_in[14];
    const float* Wh_h1  = (const float*)d_in[15];
    const float* bh_h1  = (const float*)d_in[16];
    float* out = (float*)d_out;

    const int T = 512, B = 64, H = 1024, Cin = 512;
    const int BH = B * H;        // 65536
    const int M  = T * B;        // 32768
    const int N3 = 3 * H;        // 3072
    const int Nrz = 2 * H;       // 2048

    // ---- workspace carve (all 256B aligned) ----
    char* w = (char*)d_ws;
    auto carve = [&](size_t bytes) -> char* {
        char* p = w; w += (bytes + 255) & ~(size_t)255; return p;
    };
    bf16_t* XW     = (bf16_t*)carve((size_t)M * N3 * sizeof(bf16_t));       // 192 MB, reused per layer
    bf16_t* H0bf   = (bf16_t*)carve((size_t)(T + 1) * BH * sizeof(bf16_t)); // layer0 h history (slot0 = 0)
    bf16_t* Xbf    = (bf16_t*)carve((size_t)M * Cin * sizeof(bf16_t));
    bf16_t* WcatT0 = (bf16_t*)carve((size_t)N3 * Cin * sizeof(bf16_t));     // [Wx_rz0|Wx_h0]^T
    bf16_t* WcatT1 = (bf16_t*)carve((size_t)N3 * H * sizeof(bf16_t));       // [Wx_rz1|Wx_h1]^T
    bf16_t* WArzT0 = (bf16_t*)carve((size_t)Nrz * H * sizeof(bf16_t));      // Wh_rz0^T (L2-resident)
    bf16_t* WBhT0  = (bf16_t*)carve((size_t)H * H * sizeof(bf16_t));        // Wh_h0^T
    bf16_t* WArzT1 = (bf16_t*)carve((size_t)Nrz * H * sizeof(bf16_t));
    bf16_t* WBhT1  = (bf16_t*)carve((size_t)H * H * sizeof(bf16_t));
    bf16_t* H1bf   = (bf16_t*)carve((size_t)2 * BH * sizeof(bf16_t));       // ping-pong
    bf16_t* hrbf   = (bf16_t*)carve((size_t)BH * sizeof(bf16_t));
    float*  h0f    = (float*)carve((size_t)BH * sizeof(float));
    float*  h1f    = (float*)carve((size_t)BH * sizeof(float));
    float*  zbuf   = (float*)carve((size_t)BH * sizeof(float));

    // ---- weight transpose + bf16 conversion ----
    auto tlaunch = [&](const float* in, bf16_t* op, int K, int N) {
        size_t total = (size_t)K * N;
        transpose_to_bf16<<<(unsigned)((total + 255) / 256), 256, 0, stream>>>(in, op, K, N);
    };
    tlaunch(Wx_rz0, WcatT0, Cin, Nrz);
    tlaunch(Wx_h0,  WcatT0 + (size_t)Nrz * Cin, Cin, H);
    tlaunch(Wh_rz0, WArzT0, H, Nrz);
    tlaunch(Wh_h0,  WBhT0,  H, H);
    tlaunch(Wx_rz1, WcatT1, H, Nrz);
    tlaunch(Wx_h1,  WcatT1 + (size_t)Nrz * H, H, H);
    tlaunch(Wh_rz1, WArzT1, H, Nrz);
    tlaunch(Wh_h1,  WBhT1,  H, H);

    {   size_t n = (size_t)M * Cin;
        cvt_to_bf16<<<(unsigned)((n + 255) / 256), 256, 0, stream>>>(X, Xbf, n); }
    zero_bf16<<<(BH + 255) / 256, 256, 0, stream>>>(H0bf, (size_t)BH);
    zero_bf16<<<(BH + 255) / 256, 256, 0, stream>>>(H1bf, (size_t)BH);
    zero_f32 <<<(BH + 255) / 256, 256, 0, stream>>>(h0f, (size_t)BH);
    zero_f32 <<<(BH + 255) / 256, 256, 0, stream>>>(h1f, (size_t)BH);

    // ---- layer 0: hoisted input GEMM, then sequential scan ----
    gemm_bf16_kernel<<<dim3(N3 / 64, M / 64), 128, 0, stream>>>(Xbf, Cin, WcatT0, Cin, XW, N3);
    for (int t = 0; t < T; ++t) {
        const bf16_t* XWt = XW + (size_t)t * B * N3;
        gru_step_rz<<<Nrz / 64, 128, 0, stream>>>(H0bf + (size_t)t * BH, WArzT0, XWt,
                                                  bx_rz0, bh_rz0, h0f, zbuf, hrbf);
        gru_step_out<<<H / 64, 128, 0, stream>>>(hrbf, WBhT0, XWt, bx_h0, bh_h0, zbuf,
                                                 h0f, H0bf + (size_t)(t + 1) * BH, nullptr);
    }

    // ---- layer 1: hoisted GEMM over layer-0 outputs, then scan (emits d_out rows) ----
    gemm_bf16_kernel<<<dim3(N3 / 64, M / 64), 128, 0, stream>>>(H0bf + BH, H, WcatT1, H, XW, N3);
    for (int t = 0; t < T; ++t) {
        const bf16_t* XWt = XW + (size_t)t * B * N3;
        bf16_t* rd = H1bf + (size_t)(t & 1) * BH;
        bf16_t* wr = H1bf + (size_t)((t & 1) ^ 1) * BH;
        gru_step_rz<<<Nrz / 64, 128, 0, stream>>>(rd, WArzT1, XWt,
                                                  bx_rz1, bh_rz1, h1f, zbuf, hrbf);
        gru_step_out<<<H / 64, 128, 0, stream>>>(hrbf, WBhT1, XWt, bx_h1, bh_h1, zbuf,
                                                 h1f, wr, out + (size_t)t * BH);
    }

    // ---- h_n tail: (2, B, H) after the (T, B, H) output ----
    copy_f32<<<(BH + 255) / 256, 256, 0, stream>>>(h0f, out + (size_t)T * BH, (size_t)BH);
    copy_f32<<<(BH + 255) / 256, 256, 0, stream>>>(h1f, out + (size_t)T * BH + BH, (size_t)BH);
}
